// SwinTransformerBlock3D_36472862278073
// MI455X (gfx1250) — matrix operand order repaired and checked by reading, not compiled, not run on hardware
//
#include <hip/hip_runtime.h>
#include <hip/hip_bf16.h>

typedef __attribute__((ext_vector_type(16))) _Float16 v16h;
typedef __attribute__((ext_vector_type(8)))  float    v8f;

#define HEADS 6
#define NWINDOWS 2048     // B(2) * 4 * 16 * 16

// workspace layout (bytes)
#define OFF_QKVW  0        // f16 [288][96]  (transposed qkv_w)
#define OFF_PROJW 55296    // f16 [96][96]
#define OFF_W1    73728    // f16 [384][96]
#define OFF_W2    147456   // f16 [96][384]
#define OFF_BIAS  221184   // f32 [6][64][64]
#define PREP_ELEMS (27648 + 9216 + 36864 + 36864 + 24576)  // 135168

// ---- WMMA fragment K maps (per CDNA5 ISA 7.12.2) ----
__device__ __forceinline__ int kmapA(int e, int h) { return ((e >> 3) << 4) + (h << 3) + (e & 7); }
__device__ __forceinline__ int kmapB(int e, int h) { return ((e >> 3) << 4) + (h << 3) + (e & 7); }

// A fragment: row-major f16 source [rows][ld]
__device__ __forceinline__ v16h fragA(const _Float16* base, int ld, int row0, int col0, int kvalid) {
  int l = threadIdx.x & 31, r = row0 + (l & 15), hh = l >> 4;
  v16h f;
#pragma unroll
  for (int e = 0; e < 16; ++e) {
    int kk = kmapA(e, hh);
    f[e] = (kk < kvalid) ? base[r * ld + col0 + kk] : (_Float16)0.0f;
  }
  return f;
}

// B fragment from [N][K] row-major storage
__device__ __forceinline__ v16h fragB(const _Float16* base, int ld, int row0, int col0, int kvalid) {
  int l = threadIdx.x & 31, r = row0 + (l & 15), hh = l >> 4;
  v16h f;
#pragma unroll
  for (int e = 0; e < 16; ++e) {
    int kk = kmapB(e, hh);
    f[e] = (kk < kvalid) ? base[r * ld + col0 + kk] : (_Float16)0.0f;
  }
  return f;
}

__device__ __forceinline__ v8f wmma_acc(v16h a, v16h b, v8f c) {
  return __builtin_amdgcn_wmma_f32_16x16x32_f16(false, a, false, b, (short)0, c, false, false);
}

// async 16B global->LDS copy (ASYNCcnt path); saddr form, 32-bit vaddr offset
__device__ __forceinline__ void async_copy16(unsigned lds_off, unsigned goff, unsigned long long base) {
  asm volatile("global_load_async_to_lds_b128 %0, %1, %2"
               :: "v"(lds_off), "v"(goff), "s"(base) : "memory");
}
__device__ __forceinline__ void async_wait0() {
  asm volatile("s_wait_asynccnt 0" ::: "memory");
}

union Pack8h { _Float16 h[8]; uint4 u; };
union Pack4h { _Float16 h[4]; uint2 u; };

// ================= prep: weight f16-transpose + dense rel-pos bias =================
__global__ __launch_bounds__(256) void prep_kernel(
    const float* __restrict__ qkv_w, const float* __restrict__ proj_w,
    const float* __restrict__ mlp_w1, const float* __restrict__ mlp_w2,
    const float* __restrict__ bias_table, char* __restrict__ ws) {
  int idx = blockIdx.x * 256 + threadIdx.x;
  if (idx < 27648) {                                 // qkv_wt [288][96]
    int n = idx / 96, k = idx - n * 96;
    ((_Float16*)(ws + OFF_QKVW))[idx] = (_Float16)qkv_w[k * 288 + n];
    return;
  }
  idx -= 27648;
  if (idx < 9216) {                                  // proj_wt [96][96]
    int n = idx / 96, k = idx - n * 96;
    ((_Float16*)(ws + OFF_PROJW))[idx] = (_Float16)proj_w[k * 96 + n];
    return;
  }
  idx -= 9216;
  if (idx < 36864) {                                 // w1t [384][96]
    int n = idx / 96, k = idx - n * 96;
    ((_Float16*)(ws + OFF_W1))[idx] = (_Float16)mlp_w1[k * 384 + n];
    return;
  }
  idx -= 36864;
  if (idx < 36864) {                                 // w2t [96][384]
    int n = idx / 384, k = idx - n * 384;
    ((_Float16*)(ws + OFF_W2))[idx] = (_Float16)mlp_w2[k * 96 + n];
    return;
  }
  idx -= 36864;
  if (idx < 24576) {                                 // bias [6][64][64]
    int h = idx / 4096, r = idx - h * 4096;
    int i = r / 64, j = r - i * 64;
    int ci0 = i >> 4, ci1 = (i >> 2) & 3, ci2 = i & 3;
    int cj0 = j >> 4, cj1 = (j >> 2) & 3, cj2 = j & 3;
    int t = (ci0 - cj0 + 3) * 49 + (ci1 - cj1 + 3) * 7 + (ci2 - cj2 + 3);
    ((float*)(ws + OFF_BIAS))[idx] = bias_table[t * HEADS + h];
  }
}

// shared LayerNorm helper: 4 lanes per token, quad shuffle reduction
__device__ __forceinline__ void ln_to_f16(const float* xf32, _Float16* xln,
                                          const float* g, const float* bb, int tid) {
  int tok = tid >> 2, sub = tid & 3;
  const float4* r4 = (const float4*)(xf32 + tok * 96 + sub * 24);
  float4 v[6];
#pragma unroll
  for (int c = 0; c < 6; ++c) v[c] = r4[c];
  float s = 0.f;
#pragma unroll
  for (int c = 0; c < 6; ++c) s += v[c].x + v[c].y + v[c].z + v[c].w;
  s += __shfl_xor(s, 1);
  s += __shfl_xor(s, 2);
  float m = s * (1.0f / 96.0f);
  float var = 0.f;
#pragma unroll
  for (int c = 0; c < 6; ++c) {
    float a0 = v[c].x - m, a1 = v[c].y - m, a2 = v[c].z - m, a3 = v[c].w - m;
    var += a0 * a0 + a1 * a1 + a2 * a2 + a3 * a3;
  }
  var += __shfl_xor(var, 1);
  var += __shfl_xor(var, 2);
  float rs = rsqrtf(var * (1.0f / 96.0f) + 1e-5f);
  const float4* gp = (const float4*)(g + sub * 24);
  const float4* bp = (const float4*)(bb + sub * 24);
  _Float16* o = xln + tok * 96 + sub * 24;
#pragma unroll
  for (int c = 0; c < 6; ++c) {
    float4 gg = gp[c], bbv = bp[c];
    Pack4h p;
    p.h[0] = (_Float16)((v[c].x - m) * rs * gg.x + bbv.x);
    p.h[1] = (_Float16)((v[c].y - m) * rs * gg.y + bbv.y);
    p.h[2] = (_Float16)((v[c].z - m) * rs * gg.z + bbv.z);
    p.h[3] = (_Float16)((v[c].w - m) * rs * gg.w + bbv.w);
    *(uint2*)&o[c * 4] = p.u;
  }
}

// ================= kernel A: LN1 + windowed attention + proj + residual =================
// LDS: xf32 24576 | xln 12288 | qk 24576 | vT 12288 | attn 98304 | o 12288 | srcIdx 256
#define A_LDS_BYTES (24576 + 12288 + 24576 + 12288 + 98304 + 12288 + 256)

__global__ __launch_bounds__(256) void swin_attn_kernel(
    const float* __restrict__ x, const float* __restrict__ n1g, const float* __restrict__ n1b,
    const float* __restrict__ qkv_b, const float* __restrict__ proj_b,
    const char* __restrict__ ws, float* __restrict__ out) {
  extern __shared__ char smem[];
  float*    xf32  = (float*)smem;                              // [64][96]
  _Float16* xln   = (_Float16*)(smem + 24576);                 // [64][96]
  _Float16* qk    = (_Float16*)(smem + 36864);                 // [64][192] q|k
  _Float16* vT    = (_Float16*)(smem + 61440);                 // [96][64]  v transposed
  float*    attnB = (float*)(smem + 73728);                    // [6][64][64] f32 / f16 reuse
  _Float16* oBuf  = (_Float16*)(smem + 172032);                // [64][96]
  int*      srcIx = (int*)(smem + 184320);

  const _Float16* qkv_wt  = (const _Float16*)(ws + OFF_QKVW);
  const _Float16* proj_wt = (const _Float16*)(ws + OFF_PROJW);
  const float*    biasT   = (const float*)(ws + OFF_BIAS);

  const int tid = threadIdx.x, lane = tid & 31, wid = tid >> 5, half = lane >> 4;
  const int w = blockIdx.x;
  const int b = w >> 10, rem = w & 1023;
  const int bI = rem >> 8, bX = (rem >> 4) & 15, bT = rem & 15;

  if (tid < 64) {   // rolled gather index: h[p] = x[(p + shift) mod n]
    int li = tid >> 4, lx = (tid >> 2) & 3, lt = tid & 3;
    int gI = (bI * 4 + li + 2) & 15;
    int gX = (bX * 4 + lx + 2) & 63;
    int gT = (bT * 4 + lt + 2) & 63;
    srcIx[tid] = ((b * 16 + gI) * 64 + gX) * 64 + gT;
  }
  __syncthreads();

  // async gather: 1536 chunks of 16B -> xf32
  {
    unsigned lds0 = (unsigned)(size_t)smem;
    unsigned long long xbase = (unsigned long long)(size_t)x;
#pragma unroll
    for (int it = 0; it < 6; ++it) {
      int c = tid + it * 256;
      int n = c / 24, wi = c - n * 24;
      unsigned goff = (unsigned)srcIx[n] * 384u + (unsigned)wi * 16u;
      async_copy16(lds0 + (unsigned)c * 16u, goff, xbase);
    }
    async_wait0();
  }
  __syncthreads();

  ln_to_f16(xf32, xln, n1g, n1b, tid);
  __syncthreads();

  // ---- QKV GEMM (transposed form): M=288 out-channels, N=64 tokens ----
  // q/k tiles (channels 0..191): 12 x 4 = 48 jobs, contiguous b128 store
  for (int t = wid; t < 48; t += 8) {
    int ct = t >> 2, tt = t & 3;
    int c0 = ct * 16, tok0 = tt * 16;
    v8f acc = {};
    for (int kt = 0; kt < 3; ++kt) {
      v16h a  = fragA(qkv_wt, 96, c0, kt * 32, 32);
      v16h bf = fragB(xln, 96, tok0, kt * 32, 32);
      acc = wmma_acc(a, bf, acc);
    }
    int tok = tok0 + (lane & 15);
    int cb = c0 + 8 * half;
    float4 b0 = *(const float4*)(qkv_b + cb);
    float4 b1v = *(const float4*)(qkv_b + cb + 4);
    Pack8h p;
    p.h[0] = (_Float16)(acc[0] + b0.x); p.h[1] = (_Float16)(acc[1] + b0.y);
    p.h[2] = (_Float16)(acc[2] + b0.z); p.h[3] = (_Float16)(acc[3] + b0.w);
    p.h[4] = (_Float16)(acc[4] + b1v.x); p.h[5] = (_Float16)(acc[5] + b1v.y);
    p.h[6] = (_Float16)(acc[6] + b1v.z); p.h[7] = (_Float16)(acc[7] + b1v.w);
    *(uint4*)&qk[tok * 192 + cb] = p.u;
  }
  // v tiles (channels 192..287): 6 x 4 = 24 jobs, transpose scatter into vT
  for (int t = wid; t < 24; t += 8) {
    int ct = 12 + (t >> 2), tt = t & 3;
    int c0 = ct * 16, tok0 = tt * 16;
    v8f acc = {};
    for (int kt = 0; kt < 3; ++kt) {
      v16h a  = fragA(qkv_wt, 96, c0, kt * 32, 32);
      v16h bf = fragB(xln, 96, tok0, kt * 32, 32);
      acc = wmma_acc(a, bf, acc);
    }
    int tok = tok0 + (lane & 15);
    int cb = c0 + 8 * half;
    float4 b0 = *(const float4*)(qkv_b + cb);
    float4 b1v = *(const float4*)(qkv_b + cb + 4);
    Pack8h p;
    p.h[0] = (_Float16)(acc[0] + b0.x); p.h[1] = (_Float16)(acc[1] + b0.y);
    p.h[2] = (_Float16)(acc[2] + b0.z); p.h[3] = (_Float16)(acc[3] + b0.w);
    p.h[4] = (_Float16)(acc[4] + b1v.x); p.h[5] = (_Float16)(acc[5] + b1v.y);
    p.h[6] = (_Float16)(acc[6] + b1v.z); p.h[7] = (_Float16)(acc[7] + b1v.w);
#pragma unroll
    for (int rr = 0; rr < 8; ++rr) vT[(cb - 192 + rr) * 64 + tok] = p.h[rr];
  }
  __syncthreads();

  // ---- QK^T (A=K tiles, B=Q tiles): D per lane = 8 consecutive j for one i ----
  for (int jb = wid; jb < 24; jb += 8) {
    int h = jb >> 2, jt = jb & 3;
    float* A = attnB + h * 4096;
    for (int it = 0; it < 4; ++it) {
      v16h kf = fragA(qk, 192, jt * 16, 96 + h * 16, 16);   // K padded 16->32
      v16h qf = fragB(qk, 192, it * 16, h * 16, 16);
      v8f acc = {};
      acc = wmma_acc(kf, qf, acc);
      int i = it * 16 + (lane & 15);
      int jb0 = jt * 16 + 8 * half;
      const float4* bp = (const float4*)(biasT + h * 4096 + i * 64 + jb0);
      float4 bb0 = bp[0], bb1 = bp[1];
      float4 s0, s1;
      s0.x = acc[0] * 0.25f + bb0.x; s0.y = acc[1] * 0.25f + bb0.y;
      s0.z = acc[2] * 0.25f + bb0.z; s0.w = acc[3] * 0.25f + bb0.w;
      s1.x = acc[4] * 0.25f + bb1.x; s1.y = acc[5] * 0.25f + bb1.y;
      s1.z = acc[6] * 0.25f + bb1.z; s1.w = acc[7] * 0.25f + bb1.w;
      float4* op = (float4*)(A + i * 64 + jb0);
      op[0] = s0; op[1] = s1;
    }
  }
  __syncthreads();

  // ---- softmax: 384 rows; read f32, write normalized f16 in-place (pitch 128h) ----
  for (int row = tid; row < 384; row += 256) {
    float* pr = attnB + row * 64;
    float4 vb[16];
    float mx = -3.0e38f;
#pragma unroll
    for (int j = 0; j < 16; ++j) {
      vb[j] = ((float4*)pr)[j];
      mx = fmaxf(mx, fmaxf(fmaxf(vb[j].x, vb[j].y), fmaxf(vb[j].z, vb[j].w)));
    }
    float s = 0.f;
#pragma unroll
    for (int j = 0; j < 16; ++j) {
      vb[j].x = __expf(vb[j].x - mx); vb[j].y = __expf(vb[j].y - mx);
      vb[j].z = __expf(vb[j].z - mx); vb[j].w = __expf(vb[j].w - mx);
      s += vb[j].x + vb[j].y + vb[j].z + vb[j].w;
    }
    float inv = 1.0f / s;
    _Float16* ph = (_Float16*)pr;
#pragma unroll
    for (int j = 0; j < 16; ++j) {
      Pack4h p;
      p.h[0] = (_Float16)(vb[j].x * inv); p.h[1] = (_Float16)(vb[j].y * inv);
      p.h[2] = (_Float16)(vb[j].z * inv); p.h[3] = (_Float16)(vb[j].w * inv);
      *(uint2*)&ph[j * 4] = p.u;
    }
  }
  __syncthreads();

  // ---- O = attn @ V : A=attn f16 (pitch 128), B=vT rows=channels ----
  for (int jb = wid; jb < 24; jb += 8) {
    int h = jb >> 2, mt = jb & 3;
    const _Float16* attnH = (const _Float16*)(attnB + h * 4096);
    v8f acc = {};
    for (int kt = 0; kt < 2; ++kt) {
      v16h af = fragA(attnH, 128, mt * 16, kt * 32, 32);
      v16h vf = fragB(vT, 64, h * 16, kt * 32, 32);
      acc = wmma_acc(af, vf, acc);
    }
    int d = lane & 15;
#pragma unroll
    for (int rr = 0; rr < 8; ++rr)
      oBuf[(mt * 16 + rr + 8 * half) * 96 + h * 16 + d] = (_Float16)acc[rr];
  }
  __syncthreads();

  // ---- proj (transposed form) + residual: coalesced 2x float4 stores ----
  for (int t = wid; t < 24; t += 8) {
    int nt = t / 4, mt = t & 3;          // nt: out-channel tile (6), mt: token tile (4)
    int n0 = nt * 16, m0 = mt * 16;
    v8f acc = {};
    for (int kt = 0; kt < 3; ++kt) {
      v16h a  = fragA(proj_wt, 96, n0, kt * 32, 32);
      v16h bf = fragB(oBuf, 96, m0, kt * 32, 32);
      acc = wmma_acc(a, bf, acc);
    }
    int m = m0 + (lane & 15);
    int nb = n0 + 8 * half;
    float4 pb0 = *(const float4*)(proj_b + nb);
    float4 pb1 = *(const float4*)(proj_b + nb + 4);
    const float4* xr = (const float4*)(xf32 + m * 96 + nb);
    float4 r0 = xr[0], r1 = xr[1];
    float4 s0, s1;
    s0.x = r0.x + acc[0] + pb0.x; s0.y = r0.y + acc[1] + pb0.y;
    s0.z = r0.z + acc[2] + pb0.z; s0.w = r0.w + acc[3] + pb0.w;
    s1.x = r1.x + acc[4] + pb1.x; s1.y = r1.y + acc[5] + pb1.y;
    s1.z = r1.z + acc[6] + pb1.z; s1.w = r1.w + acc[7] + pb1.w;
    float4* op = (float4*)(out + (size_t)srcIx[m] * 96 + nb);
    op[0] = s0; op[1] = s1;
  }
}

// ================= kernel B: LN2 + MLP (in-place on out) =================
#define B_LDS_BYTES (24576 + 12288 + 49152)

__global__ __launch_bounds__(256) void swin_mlp_kernel(
    const float* __restrict__ n2g, const float* __restrict__ n2b,
    const float* __restrict__ b1, const float* __restrict__ b2,
    const char* __restrict__ ws, float* __restrict__ out) {
  extern __shared__ char smem[];
  float*    xf32 = (float*)smem;                       // [64][96]
  _Float16* xln  = (_Float16*)(smem + 24576);          // [64][96]
  _Float16* hid  = (_Float16*)(smem + 36864);          // [64][384]
  const _Float16* w1t = (const _Float16*)(ws + OFF_W1);
  const _Float16* w2t = (const _Float16*)(ws + OFF_W2);

  const int tid = threadIdx.x, lane = tid & 31, wid = tid >> 5, half = lane >> 4;
  const size_t row0 = (size_t)blockIdx.x * 64;

  // async load 24KB contiguous tile -> xf32
  {
    unsigned lds0 = (unsigned)(size_t)smem;
    unsigned long long base = (unsigned long long)(size_t)(out + row0 * 96);
#pragma unroll
    for (int it = 0; it < 6; ++it) {
      int c = tid + it * 256;
      async_copy16(lds0 + (unsigned)c * 16u, (unsigned)c * 16u, base);
    }
    async_wait0();
  }
  __syncthreads();

  ln_to_f16(xf32, xln, n2g, n2b, tid);
  __syncthreads();

  // GEMM1 (transposed form) + exact GELU: M=384 hidden, N=64 tokens (24x4 tiles)
  for (int t = wid; t < 96; t += 8) {
    int ct = t >> 2, tt = t & 3;
    int c0 = ct * 16, m0 = tt * 16;
    v8f acc = {};
    for (int kt = 0; kt < 3; ++kt) {
      v16h a  = fragA(w1t, 96, c0, kt * 32, 32);
      v16h bf = fragB(xln, 96, m0, kt * 32, 32);
      acc = wmma_acc(a, bf, acc);
    }
    int m = m0 + (lane & 15);
    int cb = c0 + 8 * half;
    float4 bb0 = *(const float4*)(b1 + cb);
    float4 bb1 = *(const float4*)(b1 + cb + 4);
    float vv[8] = {acc[0] + bb0.x, acc[1] + bb0.y, acc[2] + bb0.z, acc[3] + bb0.w,
                   acc[4] + bb1.x, acc[5] + bb1.y, acc[6] + bb1.z, acc[7] + bb1.w};
    Pack8h p;
#pragma unroll
    for (int rr = 0; rr < 8; ++rr) {
      float g = 0.5f * vv[rr] * (1.0f + erff(vv[rr] * 0.70710678118f));
      p.h[rr] = (_Float16)g;
    }
    *(uint4*)&hid[m * 384 + cb] = p.u;
  }
  __syncthreads();

  // GEMM2 (transposed form) + residual: M=96 out, N=64 tokens (6x4 tiles), K=12
  for (int t = wid; t < 24; t += 8) {
    int nt = t / 4, mt = t & 3;
    int n0 = nt * 16, m0 = mt * 16;
    v8f acc = {};
    for (int kt = 0; kt < 12; ++kt) {
      v16h a  = fragA(w2t, 384, n0, kt * 32, 32);
      v16h bf = fragB(hid, 384, m0, kt * 32, 32);
      acc = wmma_acc(a, bf, acc);
    }
    int m = m0 + (lane & 15);
    int nb = n0 + 8 * half;
    float4 bb0 = *(const float4*)(b2 + nb);
    float4 bb1 = *(const float4*)(b2 + nb + 4);
    const float4* xr = (const float4*)(xf32 + m * 96 + nb);
    float4 r0 = xr[0], r1 = xr[1];
    float4 s0, s1;
    s0.x = r0.x + acc[0] + bb0.x; s0.y = r0.y + acc[1] + bb0.y;
    s0.z = r0.z + acc[2] + bb0.z; s0.w = r0.w + acc[3] + bb0.w;
    s1.x = r1.x + acc[4] + bb1.x; s1.y = r1.y + acc[5] + bb1.y;
    s1.z = r1.z + acc[6] + bb1.z; s1.w = r1.w + acc[7] + bb1.w;
    float4* op = (float4*)(out + (row0 + m) * 96 + nb);
    op[0] = s0; op[1] = s1;
  }
}

extern "C" void kernel_launch(void* const* d_in, const int* in_sizes, int n_in,
                              void* d_out, int out_size, void* d_ws, size_t ws_size,
                              hipStream_t stream) {
  const float* x          = (const float*)d_in[0];
  const float* norm1_g    = (const float*)d_in[1];
  const float* norm1_b    = (const float*)d_in[2];
  const float* qkv_w      = (const float*)d_in[3];
  const float* qkv_b      = (const float*)d_in[4];
  const float* proj_w     = (const float*)d_in[5];
  const float* proj_b     = (const float*)d_in[6];
  const float* bias_table = (const float*)d_in[7];
  const float* norm2_g    = (const float*)d_in[8];
  const float* norm2_b    = (const float*)d_in[9];
  const float* mlp_w1     = (const float*)d_in[10];
  const float* mlp_b1     = (const float*)d_in[11];
  const float* mlp_w2     = (const float*)d_in[12];
  const float* mlp_b2     = (const float*)d_in[13];
  char*  ws  = (char*)d_ws;
  float* out = (float*)d_out;

  prep_kernel<<<(PREP_ELEMS + 255) / 256, 256, 0, stream>>>(
      qkv_w, proj_w, mlp_w1, mlp_w2, bias_table, ws);
  swin_attn_kernel<<<NWINDOWS, 256, A_LDS_BYTES, stream>>>(
      x, norm1_g, norm1_b, qkv_b, proj_b, ws, out);
  swin_mlp_kernel<<<NWINDOWS, 256, B_LDS_BYTES, stream>>>(
      norm2_g, norm2_b, mlp_b1, mlp_b2, ws, out);
}
